// ScaleModel_73813307949237
// MI455X (gfx1250) — matrix operand order repaired and checked by reading, not compiled, not run on hardware
//
#include <hip/hip_runtime.h>
#include <hip/hip_bf16.h>

// ---------------------------------------------------------------------------
// Sizes from the reference: B=64, L=2048, H=128, V=32000
// ---------------------------------------------------------------------------
constexpr int BB = 64;
constexpr int LL = 2048;
constexpr int HH = 128;
constexpr int VV = 32000;

typedef __attribute__((ext_vector_type(16))) __bf16 v16bf;
typedef __attribute__((ext_vector_type(8)))  __bf16 v8bf;
typedef __attribute__((ext_vector_type(4)))  __bf16 v4bf;
typedef __attribute__((ext_vector_type(8)))  float  v8f;

// f32 -> bf16, round-to-nearest-even (done ONCE per element, not per use)
__device__ __forceinline__ __bf16 f2bf(float f) {
    unsigned u = __builtin_bit_cast(unsigned, f);
    unsigned r = (u + 0x7FFFu + ((u >> 16) & 1u)) >> 16;
    return __builtin_bit_cast(__bf16, (unsigned short)r);
}

// ---------------------------------------------------------------------------
// WMMA fragment loaders from bf16 memory (this code's contiguous k order, not the ISA table order; wave32)
// A (16x32): lane group 0 -> row M=lane, [0..15]=K(0..15);
//   lane group 1 -> K(16..31).
// ---------------------------------------------------------------------------
__device__ __forceinline__ v16bf load_a_bf(const __bf16* base, int ld, int lane) {
    const int m    = lane & 15;
    const int koff = (lane >> 4) << 4;            // 0 or 16
    const __bf16* p = base + m * ld + koff;
    v8bf lo = *(const v8bf*)(p);                  // K koff..koff+7
    v8bf hi = *(const v8bf*)(p + 8);              // K koff+8..koff+15
    return __builtin_shufflevector(lo, hi, 0, 1, 2, 3, 4, 5, 6, 7,
                                           8, 9, 10, 11, 12, 13, 14, 15);
}

// B (32x16) for X @ W^T with W row-major bf16 [N x ld]: lane = col N,
// lane group picks K 0..15 / 16..31 -> 16 contiguous bf16 = 32 bytes.
__device__ __forceinline__ v16bf load_b_bf(const __bf16* base, int ld, int lane) {
    const int n    = lane & 15;
    const int koff = (lane >> 4) << 4;            // 0 or 16
    const __bf16* p = base + n * ld + koff;
    v8bf lo = *(const v8bf*)(p);
    v8bf hi = *(const v8bf*)(p + 8);
    return __builtin_shufflevector(lo, hi, 0, 1, 2, 3, 4, 5, 6, 7,
                                           8, 9, 10, 11, 12, 13, 14, 15);
}

__device__ __forceinline__ v8f wmma_bf16(v16bf a, v16bf b, v8f c) {
    return __builtin_amdgcn_wmma_f32_16x16x32_bf16(
        false, a, false, b, (short)0, c, false, false);
}

// C/D (16x16 f32): reg r -> row M = ((lane>>4)<<3)+r, col N = lane&15.

// ---------------------------------------------------------------------------
// Prep: one-shot f32 -> bf16 weight conversion
// ---------------------------------------------------------------------------
__global__ __launch_bounds__(256)
void cvt_bf16_kernel(const float* __restrict__ src, __bf16* __restrict__ dst, int n4) {
    int idx = blockIdx.x * blockDim.x + threadIdx.x;
    if (idx >= n4) return;
    float4 v = ((const float4*)src)[idx];
    v4bf o;
    o[0] = f2bf(v.x); o[1] = f2bf(v.y); o[2] = f2bf(v.z); o[3] = f2bf(v.w);
    ((v4bf*)dst)[idx] = o;
}

// ---------------------------------------------------------------------------
// Kernel A: one 16-token tile per block, 4 waves cooperating.
// A-fragments are hoisted to registers once per stage and reused across all
// N-tiles (LDS loads were not CSE'd by the compiler across the n0 loop).
// ---------------------------------------------------------------------------
__global__ __launch_bounds__(128)
void fused_token_kernel(const int* __restrict__ seq,
                        const float* __restrict__ embed,
                        const float* __restrict__ b1, const float* __restrict__ b2,
                        const float* __restrict__ gamma, const float* __restrict__ beta,
                        const __bf16* __restrict__ w1bf, const __bf16* __restrict__ w2bf,
                        const __bf16* __restrict__ kpbf,
                        float* __restrict__ Kbuf) {
    __shared__ float lds_h[16][HH];                       // f32 h (residual)
    __shared__ alignas(16) __bf16 lds_hbf[16][HH];        // bf16 h (GEMM A)
    __shared__ alignas(16) __bf16 lds_a1[16][2 * HH];     // bf16 relu(h@w1^T+b1)
    __shared__ float lds_x[16][HH];                       // f32 x = h + ff
    __shared__ alignas(16) __bf16 lds_hn[16][HH];         // bf16 LayerNorm(x)

    const int tid  = threadIdx.x;
    const int lane = tid & 31;
    const int wave = tid >> 5;                            // 0..3
    const long tok0 = (long)blockIdx.x * 16;
    const int cn = lane & 15, crow0 = (lane >> 4) << 3;

    // 1) gather embeddings: 8 threads/row, 16 floats each; write f32 + bf16
    {
        const int row = tid >> 3, part = tid & 7;
        const int tok = seq[tok0 + row];
        const float4* src = (const float4*)(embed + (long)tok * HH) + part * 4;
        float4* dstf = (float4*)&lds_h[row][0] + part * 4;
        v4bf*   dstb = (v4bf*)&lds_hbf[row][0] + part * 4;
        #pragma unroll
        for (int i = 0; i < 4; ++i) {
            float4 v = src[i];
            dstf[i] = v;
            v4bf o;
            o[0] = f2bf(v.x); o[1] = f2bf(v.y); o[2] = f2bf(v.z); o[3] = f2bf(v.w);
            dstb[i] = o;
        }
    }
    __builtin_prefetch(w1bf + wave * 64 * HH, 0, 3);      // global_prefetch_b8
    __syncthreads();

    // 2) a1 = relu(h @ w1^T + b1): wave w owns output cols [64w, 64w+64)
    {
        v16bf a[4];
        #pragma unroll
        for (int k = 0; k < 4; ++k)
            a[k] = load_a_bf(&lds_hbf[0][0] + k * 32, HH, lane);
        #pragma unroll
        for (int ni = 0; ni < 4; ++ni) {
            const int n0 = wave * 64 + ni * 16;
            v8f c;
            const float bias = b1[n0 + cn];
            #pragma unroll
            for (int r = 0; r < 8; ++r) c[r] = bias;
            #pragma unroll
            for (int k = 0; k < 4; ++k)
                c = wmma_bf16(a[k], load_b_bf(w1bf + (long)n0 * HH + k * 32, HH, lane), c);
            #pragma unroll
            for (int r = 0; r < 8; ++r)
                lds_a1[crow0 + r][n0 + cn] = f2bf(fmaxf(c[r], 0.0f));
        }
    }
    __syncthreads();

    // 3) x = h + (a1 @ w2^T + b2): wave w owns cols [32w, 32w+32)
    {
        v16bf a[8];
        #pragma unroll
        for (int k = 0; k < 8; ++k)
            a[k] = load_a_bf(&lds_a1[0][0] + k * 32, 2 * HH, lane);
        #pragma unroll
        for (int ni = 0; ni < 2; ++ni) {
            const int n0 = wave * 32 + ni * 16;
            v8f c;
            const float bias = b2[n0 + cn];
            #pragma unroll
            for (int r = 0; r < 8; ++r) c[r] = bias;
            #pragma unroll
            for (int k = 0; k < 8; ++k)
                c = wmma_bf16(a[k], load_b_bf(w2bf + (long)n0 * (2 * HH) + k * 32, 2 * HH, lane), c);
            #pragma unroll
            for (int r = 0; r < 8; ++r)
                lds_x[crow0 + r][n0 + cn] = c[r] + lds_h[crow0 + r][n0 + cn];
        }
    }
    __syncthreads();

    // 4) LayerNorm per row (threads 0..15, one row each) -> bf16 hn
    if (tid < 16) {
        float mu = 0.f;
        #pragma unroll 8
        for (int j = 0; j < HH; ++j) mu += lds_x[tid][j];
        mu *= (1.0f / HH);
        float var = 0.f;
        #pragma unroll 8
        for (int j = 0; j < HH; ++j) { float d = lds_x[tid][j] - mu; var += d * d; }
        var *= (1.0f / HH);
        const float rs = rsqrtf(var + 1e-5f);
        #pragma unroll 8
        for (int j = 0; j < HH; ++j)
            lds_hn[tid][j] = f2bf(gamma[j] * (lds_x[tid][j] - mu) * rs + beta[j]);
    }
    __syncthreads();

    // 5) K = hn @ kp_w^T -> f32 K buffer (scan needs precision)
    {
        v16bf a[4];
        #pragma unroll
        for (int k = 0; k < 4; ++k)
            a[k] = load_a_bf(&lds_hn[0][0] + k * 32, HH, lane);
        #pragma unroll
        for (int ni = 0; ni < 2; ++ni) {
            const int n0 = wave * 32 + ni * 16;
            v8f c;
            #pragma unroll
            for (int r = 0; r < 8; ++r) c[r] = 0.0f;
            #pragma unroll
            for (int k = 0; k < 4; ++k)
                c = wmma_bf16(a[k], load_b_bf(kpbf + (long)n0 * HH + k * 32, HH, lane), c);
            #pragma unroll
            for (int r = 0; r < 8; ++r)
                Kbuf[(tok0 + crow0 + r) * HH + n0 + cn] = c[r];
        }
    }
}

// ---------------------------------------------------------------------------
// Kernel B: delta-rule scan. One block per batch; thread i owns row i of the
// 128x128 state M in registers (32 float4). Serial over t; full f32.
// ---------------------------------------------------------------------------
__global__ __launch_bounds__(128)
void delta_scan_kernel(const float* __restrict__ Kbuf, __bf16* __restrict__ readv) {
    const int b = blockIdx.x;
    const int i = threadIdx.x;
    __shared__ float4 kk[HH / 4];

    float4 m[HH / 4];
    #pragma unroll
    for (int j = 0; j < HH / 4; ++j) m[j] = make_float4(0.f, 0.f, 0.f, 0.f);

    const float* Kb = Kbuf + (long)b * LL * HH;

    for (int t = 0; t < LL - 1; ++t) {
        if (i < HH / 4) kk[i] = ((const float4*)(Kb + (long)t * HH))[i];
        __syncthreads();
        float ss = 0.f, dt = 0.f;
        #pragma unroll
        for (int j = 0; j < HH / 4; ++j) {
            const float4 kv = kk[j];
            ss += kv.x * kv.x + kv.y * kv.y + kv.z * kv.z + kv.w * kv.w;
            dt += m[j].x * kv.x + m[j].y * kv.y + m[j].z * kv.z + m[j].w * kv.w;
        }
        const float rn   = 1.0f / fmaxf(sqrtf(ss), 1e-12f);
        const float ki   = ((const float*)kk)[i];
        const float coef = (ki - dt * rn) * rn;
        #pragma unroll
        for (int j = 0; j < HH / 4; ++j) {
            const float4 kv = kk[j];
            m[j].x += coef * kv.x;  m[j].y += coef * kv.y;
            m[j].z += coef * kv.z;  m[j].w += coef * kv.w;
        }
        __syncthreads();
    }

    if (i < HH / 4) kk[i] = ((const float4*)(Kb + (long)(LL - 1) * HH))[i];
    __syncthreads();
    float rd = 0.f;
    #pragma unroll
    for (int j = 0; j < HH / 4; ++j) {
        const float4 kv = kk[j];
        rd += m[j].x * kv.x + m[j].y * kv.y + m[j].z * kv.z + m[j].w * kv.w;
    }
    readv[b * HH + i] = f2bf(rd);           // consumed only as WMMA A-fragment
}

// ---------------------------------------------------------------------------
// Kernel C1: r2 = read @ rp_w^T + rp_b (64x128 @ 128x128), bf16 out.
// ---------------------------------------------------------------------------
__global__ __launch_bounds__(32)
void rproj_kernel(const __bf16* __restrict__ readv, const __bf16* __restrict__ rpbf,
                  const float* __restrict__ rp_b, __bf16* __restrict__ r2) {
    const int lane = threadIdx.x;
    const int m0 = blockIdx.x * 16, n0 = blockIdx.y * 16;
    const int cn = lane & 15, crow0 = (lane >> 4) << 3;

    v8f c;
    const float bias = rp_b[n0 + cn];
    #pragma unroll
    for (int r = 0; r < 8; ++r) c[r] = bias;
    #pragma unroll
    for (int k = 0; k < 4; ++k) {
        v16bf a = load_a_bf(readv + (long)m0 * HH + k * 32, HH, lane);
        v16bf b = load_b_bf(rpbf + (long)n0 * HH + k * 32, HH, lane);
        c = wmma_bf16(a, b, c);
    }
    #pragma unroll
    for (int r = 0; r < 8; ++r)
        r2[(m0 + crow0 + r) * HH + n0 + cn] = f2bf(c[r]);
}

// ---------------------------------------------------------------------------
// Kernel C2: out = r2 @ out_w^T + out_b (64x128 @ 128x32000), f32 out.
// Memory-bound on bf16 out_w (8 MB). Each wave owns 4 vocab tiles so the
// r2 A-fragments are loaded once and out_w is pure streamed B traffic.
// ---------------------------------------------------------------------------
__global__ __launch_bounds__(32)
void logits_kernel(const __bf16* __restrict__ r2, const __bf16* __restrict__ owbf,
                   const float* __restrict__ out_b, float* __restrict__ out) {
    const int lane = threadIdx.x;
    const int v00 = blockIdx.x * 64;              // 4 vocab tiles per wave
    const int m0  = blockIdx.y * 16;
    const int cn = lane & 15, crow0 = (lane >> 4) << 3;

    __builtin_prefetch(owbf + (long)v00 * HH + (lane << 3), 0, 1);

    v16bf a[4];
    #pragma unroll
    for (int k = 0; k < 4; ++k)
        a[k] = load_a_bf(r2 + (long)m0 * HH + k * 32, HH, lane);

    #pragma unroll
    for (int vt = 0; vt < 4; ++vt) {
        const int v0 = v00 + vt * 16;
        v8f c;
        const float bias = out_b[v0 + cn];
        #pragma unroll
        for (int r = 0; r < 8; ++r) c[r] = bias;
        #pragma unroll
        for (int k = 0; k < 4; ++k)
            c = wmma_bf16(a[k], load_b_bf(owbf + (long)v0 * HH + k * 32, HH, lane), c);
        #pragma unroll
        for (int r = 0; r < 8; ++r)
            out[(long)(m0 + crow0 + r) * VV + v0 + cn] = c[r];
    }
}

// ---------------------------------------------------------------------------
extern "C" void kernel_launch(void* const* d_in, const int* in_sizes, int n_in,
                              void* d_out, int out_size, void* d_ws, size_t ws_size,
                              hipStream_t stream) {
    const int*   seq   = (const int*)  d_in[0];
    const float* embed = (const float*)d_in[1];
    const float* w1    = (const float*)d_in[2];
    const float* b1    = (const float*)d_in[3];
    const float* w2    = (const float*)d_in[4];
    const float* b2    = (const float*)d_in[5];
    const float* gamma = (const float*)d_in[6];
    const float* beta  = (const float*)d_in[7];
    const float* kp_w  = (const float*)d_in[8];
    const float* rp_w  = (const float*)d_in[9];
    const float* rp_b  = (const float*)d_in[10];
    const float* out_w = (const float*)d_in[11];
    const float* out_b = (const float*)d_in[12];
    float* out = (float*)d_out;

    // ---- workspace layout (64B aligned slices) ----
    char* base = (char*)d_ws;
    size_t off = 0;
    auto take = [&](size_t bytes) -> char* {
        char* p = base + off;
        off = (off + bytes + 63) & ~(size_t)63;
        return p;
    };
    float*  Kbuf   = (float*) take((size_t)BB * LL * HH * 4);   // 64 MB
    __bf16* w1bf   = (__bf16*)take((size_t)2 * HH * HH * 2);
    __bf16* w2bf   = (__bf16*)take((size_t)2 * HH * HH * 2);
    __bf16* kpbf   = (__bf16*)take((size_t)HH * HH * 2);
    __bf16* rpbf   = (__bf16*)take((size_t)HH * HH * 2);
    __bf16* owbf   = (__bf16*)take((size_t)VV * HH * 2);        // 8 MB
    __bf16* readbf = (__bf16*)take((size_t)BB * HH * 2);
    __bf16* r2bf   = (__bf16*)take((size_t)BB * HH * 2);

    // ---- one-shot weight conversions (bf16 weights reused by all tiles) ----
    cvt_bf16_kernel<<<(2 * HH * HH / 4 + 255) / 256, 256, 0, stream>>>(w1,   w1bf, 2 * HH * HH / 4);
    cvt_bf16_kernel<<<(2 * HH * HH / 4 + 255) / 256, 256, 0, stream>>>(w2,   w2bf, 2 * HH * HH / 4);
    cvt_bf16_kernel<<<(HH * HH / 4 + 255) / 256, 256, 0, stream>>>(kp_w, kpbf, HH * HH / 4);
    cvt_bf16_kernel<<<(HH * HH / 4 + 255) / 256, 256, 0, stream>>>(rp_w, rpbf, HH * HH / 4);
    cvt_bf16_kernel<<<(VV * HH / 4 + 255) / 256, 256, 0, stream>>>(out_w, owbf, VV * HH / 4);

    // ---- pipeline ----
    fused_token_kernel<<<(BB * LL) / 16, 128, 0, stream>>>(
        seq, embed, b1, b2, gamma, beta, w1bf, w2bf, kpbf, Kbuf);

    delta_scan_kernel<<<BB, HH, 0, stream>>>(Kbuf, readbf);

    rproj_kernel<<<dim3(BB / 16, HH / 16), 32, 0, stream>>>(readbf, rpbf, rp_b, r2bf);

    logits_kernel<<<dim3(VV / 64, BB / 16), 32, 0, stream>>>(r2bf, owbf, out_b, out);
}
